// HPUAttentionImpl_43267500540271
// MI455X (gfx1250) — matrix operand order repaired and checked by reading, not compiled, not run on hardware
//
#include <hip/hip_runtime.h>

// Paged decode attention for MI455X (gfx1250, wave32, WMMA).
// B=32, H=32, H_KV=8, G=4, D=128, BS=128, NB=16, T=512. fp32 in/out,
// f16 WMMA (f32 accumulate) for QK^T and PV. HBM-bound: ~537 MB K/V read once.

#define BB   32
#define HH   32
#define HKV  8
#define GG   4
#define DD   128
#define BSZ  128
#define NBK  16
#define KVSTRIDE (HKV*DD)               // floats between consecutive positions
#define QK_SCALE 0.08838834764831845f   // 1/sqrt(128)

typedef _Float16 half16 __attribute__((ext_vector_type(16)));
typedef _Float16 half8  __attribute__((ext_vector_type(8)));
typedef float    float8 __attribute__((ext_vector_type(8)));
typedef float    f32x4  __attribute__((ext_vector_type(4)));

// 16 contiguous f32 -> 16 f16, non-temporal b128 loads (streamed once).
__device__ __forceinline__ half16 cvt_f32x16_nt(const float* p) {
  const f32x4* q = reinterpret_cast<const f32x4*>(p);
  half16 r;
#pragma unroll
  for (int i = 0; i < 4; ++i) {
    f32x4 f = __builtin_nontemporal_load(q + i);
    r[4*i+0] = (_Float16)f.x; r[4*i+1] = (_Float16)f.y;
    r[4*i+2] = (_Float16)f.z; r[4*i+3] = (_Float16)f.w;
  }
  return r;
}

__global__ __launch_bounds__(256) void paged_decode_attn(
    const float* __restrict__ query,        // [B, H*D]
    const float* __restrict__ knew,         // [B, HKV*D]
    const float* __restrict__ vnew,         // [B, HKV*D]
    const float* __restrict__ kcache,       // [NC, BS, HKV, D]
    const float* __restrict__ vcache,       // [NC, BS, HKV, D]
    const int*   __restrict__ block_list,   // [T]
    const int*   __restrict__ block_groups, // [T] (structure = t/NB)
    const int*   __restrict__ block_indices,// [B]
    const int*   __restrict__ block_offsets,// [B]
    const float* __restrict__ bias,         // [T, BS]
    float*       __restrict__ out)          // [B, H*D]
{
  __shared__ __attribute__((aligned(16))) _Float16 p_lds[8][16*128]; // per-wave P tile
  __shared__ float o_lds[8][4][128];
  __shared__ float m_lds[8][4];
  __shared__ float s_lds[8][4];

  const int bid  = blockIdx.x;
  const int seq  = bid / HKV;
  const int kv   = bid % HKV;
  const int tid  = threadIdx.x;
  const int wv   = tid >> 5;
  const int lane = tid & 31;
  const int hi   = (lane >> 4) & 1;
  const int l15  = lane & 15;

  (void)block_groups; // grouping is t/NB by construction in this setup

  // Zero padded P rows 4..15 once (they stay zero -> O rows 4..15 stay zero).
  for (int i = lane; i < 12*128; i += 32) p_lds[wv][4*128 + i] = (_Float16)0.f;

  const float* knew_row = knew + ((size_t)seq*HKV + kv)*DD;
  const float* vnew_row = vnew + ((size_t)seq*HKV + kv)*DD;

  // ---- Q A-fragments (16x32 f16 layout): row M = g (pad to 16), K-dim = d.
  // lo lanes: elems 0..7 -> K = dc*32+0..7,   elems 8..15 -> K = dc*32+8..15
  // hi lanes: elems 0..7 -> K = dc*32+16..23, elems 8..15 -> K = dc*32+24..31
  half16 qf[4];
  {
    const int g = l15;
    const float* qrow = query + ((size_t)seq*HH + (size_t)kv*GG + (size_t)g)*DD;
#pragma unroll
    for (int dc = 0; dc < 4; ++dc) {
      const int b0 = dc*32 + hi*16;
      const int b1 = dc*32 + hi*16 + 8;
#pragma unroll
      for (int e = 0; e < 8; ++e) {
        const float v0 = (g < GG) ? qrow[b0 + e] * QK_SCALE : 0.f;
        const float v1 = (g < GG) ? qrow[b1 + e] * QK_SCALE : 0.f;
        qf[dc][e]     = (_Float16)v0;
        qf[dc][8 + e] = (_Float16)v1;
      }
    }
  }

  // Online-softmax running state (rows g=0..3; valid on lanes 0..15).
  float mrun[4], srun[4];
#pragma unroll
  for (int r = 0; r < 4; ++r) { mrun[r] = -1e30f; srun[r] = 0.f; }
  float8 Oacc[8];
#pragma unroll
  for (int dt = 0; dt < 8; ++dt)
#pragma unroll
    for (int r = 0; r < 8; ++r) Oacc[dt][r] = 0.f;

  // Each wave processes 2 of the sequence's 16 mapped blocks.
  for (int jj = 0; jj < 2; ++jj) {
    const int t   = seq*NBK + wv*2 + jj;
    const int blk = block_list[t];
    const int patch_s = (blk == block_indices[seq]) ? block_offsets[seq] : -1;

    const float* kblk = kcache + ((size_t)blk*BSZ*HKV + kv)*(size_t)DD;
    const float* vblk = vcache + ((size_t)blk*BSZ*HKV + kv)*(size_t)DD;
    const float* brow = bias + (size_t)t*BSZ;

    // ---- S = (Q*scale) K^T + bias : 8 column tiles of 16 positions.
    float8 S[8];
#pragma unroll
    for (int tc = 0; tc < 8; ++tc) {
      const int s = tc*16 + l15;               // position (column N) for this lane
      const float* krow = (s == patch_s) ? knew_row : (kblk + (size_t)s*KVSTRIDE);
      float8 c;
#pragma unroll
      for (int r = 0; r < 8; ++r) c[r] = 0.f;
#pragma unroll
      for (int dc = 0; dc < 4; ++dc) {
        // B 32x16: lo lanes hold K=dc*32+0..15, hi lanes K=dc*32+16..31 (contig f32).
        half16 bf = cvt_f32x16_nt(krow + dc*32 + hi*16);
        c = __builtin_amdgcn_wmma_f32_16x16x32_f16(false, qf[dc], false, bf,
                                                   (short)0, c, false, false);
      }
      const float bv = brow[s];                // bias depends only on position
#pragma unroll
      for (int r = 0; r < 8; ++r) c[r] += bv;
      S[tc] = c;
    }

    // ---- row max over 128 positions (rows 0..3), online rescale factors.
    float fac[4], mnew_[4];
#pragma unroll
    for (int r = 0; r < 4; ++r) {
      float v = S[0][r];
#pragma unroll
      for (int tc = 1; tc < 8; ++tc) v = fmaxf(v, S[tc][r]);
      v = fmaxf(v, __shfl_xor(v, 1));
      v = fmaxf(v, __shfl_xor(v, 2));
      v = fmaxf(v, __shfl_xor(v, 4));
      v = fmaxf(v, __shfl_xor(v, 8));
      v = fmaxf(v, -1e30f);                    // fully-masked block -> no NaN
      const float mo = mrun[r];
      const float mn = fmaxf(mo, v);
      fac[r]   = expf(mo - mn);
      mnew_[r] = mn;
      mrun[r]  = mn;
    }

    // ---- P = exp(S - m), block sums, stage P to LDS (rows 0..3 only).
    float bs[4] = {0.f, 0.f, 0.f, 0.f};
    if (!hi) {
#pragma unroll
      for (int r = 0; r < 4; ++r) {
#pragma unroll
        for (int tc = 0; tc < 8; ++tc) {
          const float p = expf(S[tc][r] - mnew_[r]);
          bs[r] += p;
          p_lds[wv][r*128 + tc*16 + l15] = (_Float16)p;
        }
      }
    }
#pragma unroll
    for (int r = 0; r < 4; ++r) {
      float v = bs[r];
      v += __shfl_xor(v, 1);
      v += __shfl_xor(v, 2);
      v += __shfl_xor(v, 4);
      v += __shfl_xor(v, 8);
      srun[r] = srun[r]*fac[r] + v;
    }

    // ---- rescale O accumulator (rows >=4 are exactly zero, factor harmless).
#pragma unroll
    for (int dt = 0; dt < 8; ++dt)
#pragma unroll
      for (int r = 0; r < 8; ++r) Oacc[dt][r] *= fac[r & 3];

    __syncthreads();  // order P stores before A-fragment loads (uniform)

    // ---- O += P V : K-dim = positions (4 chunks of 32), N = d (8 tiles of 16).
    // V fragments load straight from the cache (no patch selects): base + e*4KB
    // immediate offsets -> clauses of global_load_b32.
#pragma unroll
    for (int pc = 0; pc < 4; ++pc) {
      half16 pa;                               // A-fragment of P from LDS
      {
        const int row = l15;
        const int o0 = row*128 + pc*32 + hi*16; // 16B-aligned runs of 8 halves
        half8 a0 = *reinterpret_cast<const half8*>(&p_lds[wv][o0]);
        half8 a1 = *reinterpret_cast<const half8*>(&p_lds[wv][o0 + 8]);
#pragma unroll
        for (int e = 0; e < 8; ++e) { pa[e] = a0[e]; pa[8 + e] = a1[e]; }
      }
      const float* vp0 = vblk + (size_t)(pc*32 + hi*16)*KVSTRIDE;
#pragma unroll
      for (int dt = 0; dt < 8; ++dt) {
        const float* vp = vp0 + dt*16 + l15;   // column N = d for this lane
        half16 vf;
#pragma unroll
        for (int e = 0; e < 16; ++e)           // K-dim position = pc*32+hi*16+e
          vf[e] = (_Float16)__builtin_nontemporal_load(vp + e*KVSTRIDE);
        Oacc[dt] = __builtin_amdgcn_wmma_f32_16x16x32_f16(false, pa, false, vf,
                                                          (short)0, Oacc[dt],
                                                          false, false);
      }
    }

    // ---- rank-1 patch correction: O += P[:,ps] (x) (v_new - v_old).
    // Equivalent to having scattered the new token into the V cache.
    if (patch_s >= 0) {                        // wave-uniform branch
      const float pc0 = (float)p_lds[wv][0*128 + patch_s];
      const float pc1 = (float)p_lds[wv][1*128 + patch_s];
      const float pc2 = (float)p_lds[wv][2*128 + patch_s];
      const float pc3 = (float)p_lds[wv][3*128 + patch_s];
      const float* vold = vblk + (size_t)patch_s*KVSTRIDE;
      const float msk = hi ? 0.f : 1.f;        // only rows 0..3 (lo lanes) real
#pragma unroll
      for (int dt = 0; dt < 8; ++dt) {
        const int d = dt*16 + l15;
        const float dv = (vnew_row[d] - vold[d]) * msk;
        Oacc[dt][0] += pc0 * dv;
        Oacc[dt][1] += pc1 * dv;
        Oacc[dt][2] += pc2 * dv;
        Oacc[dt][3] += pc3 * dv;
      }
    }
    __syncthreads();  // before next iteration overwrites P (uniform)
  }

  // ---- publish per-wave partials.
  if (!hi) {
#pragma unroll
    for (int r = 0; r < 4; ++r)
#pragma unroll
      for (int dt = 0; dt < 8; ++dt)
        o_lds[wv][r][dt*16 + l15] = Oacc[dt][r];
  }
  if (lane == 0) {
#pragma unroll
    for (int r = 0; r < 4; ++r) { m_lds[wv][r] = mrun[r]; s_lds[wv][r] = srun[r]; }
  }
  __syncthreads();

  // ---- merge 8 wave-partials (log-sum-exp combine) and write output.
  for (int idx = tid; idx < GG*DD; idx += 256) {
    const int g = idx >> 7;
    const int d = idx & 127;
    float M = -1e30f;
#pragma unroll
    for (int w = 0; w < 8; ++w) M = fmaxf(M, m_lds[w][g]);
    float denom = 0.f, acc = 0.f;
#pragma unroll
    for (int w = 0; w < 8; ++w) {
      const float e = expf(m_lds[w][g] - M);
      denom += s_lds[w][g] * e;
      acc   += o_lds[w][g][d] * e;
    }
    out[((size_t)seq*HH + (size_t)kv*GG + g)*DD + d] = acc / denom;
  }
}

extern "C" void kernel_launch(void* const* d_in, const int* in_sizes, int n_in,
                              void* d_out, int out_size, void* d_ws, size_t ws_size,
                              hipStream_t stream) {
  (void)in_sizes; (void)n_in; (void)out_size; (void)d_ws; (void)ws_size;
  const float* query = (const float*)d_in[0];
  const float* knew  = (const float*)d_in[1];
  const float* vnew  = (const float*)d_in[2];
  const float* kc    = (const float*)d_in[3];
  const float* vc    = (const float*)d_in[4];
  const int*   blist = (const int*)d_in[5];
  const int*   bgrp  = (const int*)d_in[6];
  const int*   bidx  = (const int*)d_in[7];
  const int*   boff  = (const int*)d_in[8];
  const float* bias  = (const float*)d_in[9];
  float* out = (float*)d_out;

  paged_decode_attn<<<dim3(BB*HKV), dim3(256), 0, stream>>>(
      query, knew, vnew, kc, vc, blist, bgrp, bidx, boff, bias, out);
}
